// Net_22986664968500
// MI455X (gfx1250) — matrix operand order repaired and checked by reading, not compiled, not run on hardware
//
#include <hip/hip_runtime.h>
#include <hip/hip_bf16.h>

typedef __attribute__((ext_vector_type(16))) __bf16 v16bf;
typedef __attribute__((ext_vector_type(8)))  float  v8f;
typedef __attribute__((ext_vector_type(4)))  float  v4f;   // ext-vector for NT loads

#define N_NODES 100000
#define N_EDGES 3200000
#define D_IN    512
#define D_HID   16
#define N_CLS   7

// ---------------------------------------------------------------------------
// Pre-pack W1[512][16] (f32) into bf16 in the exact per-lane WMMA-B layout:
//   W1bf[(c*32 + lane)*16 + i] = bf16( W1[(c*32 + (i>>3)*16 + (lane>>4)*8 + (i&7))][lane&15] )
// for K-chunk c = k0/32. 16 chunks * 32 lanes * 16 vals = 8192 bf16 = 16 KB.
// ---------------------------------------------------------------------------
__global__ void pack_w1_bf16(const float* __restrict__ W1, __bf16* __restrict__ W1bf)
{
    int tid = blockIdx.x * blockDim.x + threadIdx.x;       // 0..8191
    if (tid >= (D_IN / 32) * 32 * 16) return;
    int i    = tid & 15;
    int lane = (tid >> 4) & 31;
    int c    = tid >> 9;
    int half = lane >> 4;
    int m    = lane & 15;
    W1bf[tid] = (__bf16)W1[(c * 32 + (i >> 3) * 16 + half * 8 + (i & 7)) * D_HID + m];
}

// ---------------------------------------------------------------------------
// GEMM1: h1[N_NODES][16] = x[N_NODES][512] @ W1[512][16]   (bf16 WMMA, f32 acc)
// One wave32 per 16-row tile; 16 x v_wmma_f32_16x16x32_bf16 per tile.
// x streamed with NT hint (read-once, 204.8 MB) to keep h1/agg1 L2-resident.
// ---------------------------------------------------------------------------
__global__ __launch_bounds__(256) void gemm1_wmma(const float* __restrict__ x,
                                                  const __bf16* __restrict__ W1bf,
                                                  float* __restrict__ h1)
{
    const int wave = (int)((blockIdx.x * blockDim.x + threadIdx.x) >> 5);
    const int lane = threadIdx.x & 31;
    const int nTiles = N_NODES / 16;          // 6250, exact
    if (wave >= nTiles) return;               // uniform per wave -> EXEC stays all-1s

    const int m    = lane & 15;               // column for B/C, row for A
    const int half = lane >> 4;               // lane group 0/1
    const float* xrow = x + (size_t)(wave * 16 + m) * D_IN;

    v8f acc = {};
    #pragma unroll 2
    for (int k0 = 0; k0 < D_IN; k0 += 32) {
        // ---- A: 16-bit A-matrix 16x32 layout ----
        // low lanes need K = k0+{0..7,16..23}; high lanes K = k0+{8..15,24..31}
        const int ko = k0 + half * 8;
        const v4f c0a = __builtin_nontemporal_load((const v4f*)(xrow + ko));
        const v4f c0b = __builtin_nontemporal_load((const v4f*)(xrow + ko + 4));
        const v4f c1a = __builtin_nontemporal_load((const v4f*)(xrow + ko + 16));
        const v4f c1b = __builtin_nontemporal_load((const v4f*)(xrow + ko + 20));
        v16bf a;
        a[0]  = (__bf16)c0a.x; a[1]  = (__bf16)c0a.y; a[2]  = (__bf16)c0a.z; a[3]  = (__bf16)c0a.w;
        a[4]  = (__bf16)c0b.x; a[5]  = (__bf16)c0b.y; a[6]  = (__bf16)c0b.z; a[7]  = (__bf16)c0b.w;
        a[8]  = (__bf16)c1a.x; a[9]  = (__bf16)c1a.y; a[10] = (__bf16)c1a.z; a[11] = (__bf16)c1a.w;
        a[12] = (__bf16)c1b.x; a[13] = (__bf16)c1b.y; a[14] = (__bf16)c1b.z; a[15] = (__bf16)c1b.w;

        // ---- B: pre-packed per-lane layout, one 32B vector load ----
        const v16bf b = *(const v16bf*)(W1bf + ((((size_t)(k0 >> 5)) * 32 + lane) << 4));

        acc = __builtin_amdgcn_wmma_f32_16x16x32_bf16(
                  false, a, false, b, (short)0, acc, false, false);
    }

    // D layout: VGPR r -> row half*8 + r, col m (coalesced per half-wave)
    #pragma unroll
    for (int r = 0; r < 8; ++r)
        h1[(size_t)(wave * 16 + half * 8 + r) * D_HID + m] = acc[r];
}

// ---------------------------------------------------------------------------
__global__ void fill_zero(float* __restrict__ p, int n)
{
    int i = blockIdx.x * blockDim.x + threadIdx.x;
    if (i < n) p[i] = 0.0f;
}

__global__ void fill_bias7(float* __restrict__ out, const float* __restrict__ b2)
{
    int i = blockIdx.x * blockDim.x + threadIdx.x;
    if (i < N_NODES * N_CLS) out[i] = b2[i % N_CLS];
}

// Scatter-add of 16-wide messages: 4 threads per edge, float4 gather + 4 atomics.
// h1 (6.4 MB) and agg1 (6.4 MB) are L2-resident; atomics resolve at L2.
__global__ __launch_bounds__(256) void scatter16(const float* __restrict__ h1,
                                                 const int* __restrict__ src,
                                                 const int* __restrict__ dst,
                                                 float* __restrict__ agg)
{
    long long tid = (long long)blockIdx.x * blockDim.x + threadIdx.x;
    long long e = tid >> 2;
    int q = (int)(tid & 3);
    if (e >= N_EDGES) return;
    int s = src[e];
    int d = dst[e];
    const float4 v = *(const float4*)(h1 + (size_t)s * D_HID + q * 4);
    float* ap = agg + (size_t)d * D_HID + q * 4;
    atomicAdd(ap + 0, v.x);
    atomicAdd(ap + 1, v.y);
    atomicAdd(ap + 2, v.z);
    atomicAdd(ap + 3, v.w);
}

// Fused: h2[r][0..6] = relu(agg1[r] + b1) @ W2 ; h2 row stride 8 for alignment.
__global__ __launch_bounds__(256) void relu_bias_gemm2(const float* __restrict__ agg1,
                                                       const float* __restrict__ b1,
                                                       const float* __restrict__ W2,
                                                       float* __restrict__ h2)
{
    int r = blockIdx.x * blockDim.x + threadIdx.x;
    if (r >= N_NODES) return;
    float hr[D_HID];
    const float4* a4 = (const float4*)(agg1 + (size_t)r * D_HID);
    #pragma unroll
    for (int q = 0; q < 4; ++q) {
        float4 v = a4[q];
        hr[q*4+0] = v.x; hr[q*4+1] = v.y; hr[q*4+2] = v.z; hr[q*4+3] = v.w;
    }
    #pragma unroll
    for (int k = 0; k < D_HID; ++k) {
        float t = hr[k] + b1[k];
        hr[k] = t > 0.0f ? t : 0.0f;
    }
    float y[N_CLS];
    #pragma unroll
    for (int c = 0; c < N_CLS; ++c) y[c] = 0.0f;
    #pragma unroll
    for (int k = 0; k < D_HID; ++k) {
        float hv = hr[k];
        #pragma unroll
        for (int c = 0; c < N_CLS; ++c)
            y[c] = fmaf(hv, W2[k * N_CLS + c], y[c]);
    }
    float* o = h2 + (size_t)r * 8;
    #pragma unroll
    for (int c = 0; c < N_CLS; ++c) o[c] = y[c];
    o[7] = 0.0f;
}

// Scatter-add of 7-wide messages: thread per edge.
__global__ __launch_bounds__(256) void scatter7(const float* __restrict__ h2,
                                                const int* __restrict__ src,
                                                const int* __restrict__ dst,
                                                float* __restrict__ out)
{
    long long e = (long long)blockIdx.x * blockDim.x + threadIdx.x;
    if (e >= N_EDGES) return;
    int s = src[e];
    int d = dst[e];
    const float* hp = h2 + (size_t)s * 8;
    float* op = out + (size_t)d * N_CLS;
    #pragma unroll
    for (int c = 0; c < N_CLS; ++c)
        atomicAdd(op + c, hp[c]);
}

// ---------------------------------------------------------------------------
extern "C" void kernel_launch(void* const* d_in, const int* in_sizes, int n_in,
                              void* d_out, int out_size, void* d_ws, size_t ws_size,
                              hipStream_t stream)
{
    const float* x  = (const float*)d_in[0];
    const int*   ei = (const int*)  d_in[1];   // [2, E]: src row then dst row
    const float* W1 = (const float*)d_in[2];
    const float* b1 = (const float*)d_in[3];
    const float* W2 = (const float*)d_in[4];
    const float* b2 = (const float*)d_in[5];
    const int* src = ei;
    const int* dst = ei + N_EDGES;
    float* out = (float*)d_out;

    // Workspace layout (~16 MB total)
    float*  h1   = (float*)d_ws;                        // N_NODES*16 f32  (6.4 MB)
    float*  agg1 = h1   + (size_t)N_NODES * D_HID;      // N_NODES*16 f32  (6.4 MB)
    float*  h2   = agg1 + (size_t)N_NODES * D_HID;      // N_NODES*8  f32  (3.2 MB)
    __bf16* w1bf = (__bf16*)(h2 + (size_t)N_NODES * 8); // 8192 bf16 (16 KB, 32B-aligned)

    // 0) Pack W1 into WMMA-B bf16 layout (tiny, once per call)
    pack_w1_bf16<<<32, 256, 0, stream>>>(W1, w1bf);

    // 1) h1 = x @ W1 via bf16 WMMA (memory-bound: 204.8 MB of x @ 23.3 TB/s)
    {
        int waves   = N_NODES / 16;                     // 6250
        int threads = 256;                              // 8 waves per block
        int blocks  = (waves * 32 + threads - 1) / threads;
        gemm1_wmma<<<blocks, threads, 0, stream>>>(x, w1bf, h1);
    }
    // 2) agg1 = 0 (must re-zero every call: atomics accumulate)
    fill_zero<<<(N_NODES * D_HID + 255) / 256, 256, 0, stream>>>(agg1, N_NODES * D_HID);
    // 3) agg1 += scatter(h1[src] -> dst)
    {
        long long nth = (long long)N_EDGES * 4;
        scatter16<<<(unsigned)((nth + 255) / 256), 256, 0, stream>>>(h1, src, dst, agg1);
    }
    // 4) h2 = relu(agg1 + b1) @ W2
    relu_bias_gemm2<<<(N_NODES + 255) / 256, 256, 0, stream>>>(agg1, b1, W2, h2);
    // 5) out = b2 (broadcast), then scatter-add h2[src] -> dst
    fill_bias7<<<(N_NODES * N_CLS + 255) / 256, 256, 0, stream>>>(out, b2);
    scatter7<<<(N_EDGES + 255) / 256, 256, 0, stream>>>(h2, src, dst, out);
}
